// Switch_78735340471040
// MI455X (gfx1250) — matrix operand order repaired and checked by reading, not compiled, not run on hardware
//
#include <hip/hip_runtime.h>
#include <stdint.h>

#define NTOK   2048
#define DMODEL 1024
#define NEXP   8
#define FHID   4096

typedef __attribute__((ext_vector_type(16))) __bf16    v16bf;
typedef __attribute__((ext_vector_type(8)))  float     v8f;
typedef __attribute__((ext_vector_type(4)))  unsigned  v4u;
typedef __attribute__((ext_vector_type(8)))  int       v8i;
typedef __attribute__((ext_vector_type(4)))  int       v4i;

// ---- workspace layout (bytes) ----
#define XB_OFF   (0ul)                       // bf16 x        : 4 MB
#define W1B_OFF  (4ul << 20)                 // bf16 w1 tiles : 64 MB
#define W2B_OFF  (68ul << 20)                // bf16 w2 tiles : 64 MB
#define GATE_OFF (132ul << 20)               // f32 gate*mask : 8 KB
#define EIDX_OFF (GATE_OFF + 8192ul)         // i32 expert id : 8 KB
#define CNT_OFF  (EIDX_OFF + 8192ul)         // i32 counts    : 128 B
#define BKT_OFF  (CNT_OFF + 128ul)           // i32 buckets   : 64 KB

// f32 -> bf16 round-to-nearest-even (bit manipulation; no __bf16 scalar math)
__device__ __forceinline__ unsigned short f2bf(float f) {
  union { float f; unsigned u; } c; c.f = f;
  unsigned r = c.u + 0x7FFFu + ((c.u >> 16) & 1u);
  return (unsigned short)(r >> 16);
}

// ---------------------------------------------------------------------------
// Kernel 1: top-1 router + bf16 conversion of x. One block per token.
// ---------------------------------------------------------------------------
__global__ void __launch_bounds__(256) k_router(const float* __restrict__ x,
                                                const int* __restrict__ mask,
                                                const float* __restrict__ rw,
                                                char* __restrict__ ws) {
  __shared__ float xrow[DMODEL];
  __shared__ float logitS[NEXP];
  const int n   = blockIdx.x;
  const int tid = threadIdx.x;
  if (blockIdx.x == 0 && tid < NEXP) ((int*)(ws + CNT_OFF))[tid] = 0;

  const float* xr = x + (size_t)n * DMODEL;
  float4 v = ((const float4*)xr)[tid];           // 256*4 = 1024
  ((float4*)xrow)[tid] = v;

  uint2 p;
  p.x = (unsigned)f2bf(v.x) | ((unsigned)f2bf(v.y) << 16);
  p.y = (unsigned)f2bf(v.z) | ((unsigned)f2bf(v.w) << 16);
  *(uint2*)((unsigned short*)(ws + XB_OFF) + (size_t)n * DMODEL + tid * 4) = p;
  __syncthreads();

  const int wv = tid >> 5, lane = tid & 31;      // wave wv computes expert wv
  const float* rwe = rw + wv * DMODEL;
  float part = 0.f;
  #pragma unroll 4
  for (int i = 0; i < DMODEL / 32; ++i) {
    int k = lane + i * 32;
    part += xrow[k] * rwe[k];
  }
  for (int off = 16; off > 0; off >>= 1) part += __shfl_down(part, off, 32);
  if (lane == 0) logitS[wv] = part;
  __syncthreads();

  if (tid == 0) {
    float mx = logitS[0]; int am = 0;
    for (int e = 1; e < NEXP; ++e)
      if (logitS[e] > mx) { mx = logitS[e]; am = e; }   // first max == argmax
    float s = 0.f;
    for (int e = 0; e < NEXP; ++e) s += __expf(logitS[e] - mx);
    float gate = 1.0f / s;                              // softmax prob at argmax
    ((int*)(ws + EIDX_OFF))[n]   = am;
    ((float*)(ws + GATE_OFF))[n] = gate * (mask[n] ? 1.0f : 0.0f);
  }
}

// ---------------------------------------------------------------------------
// Kernel 2: convert w1/w2 to bf16, pre-shuffled into WMMA B-fragment order.
// B operand 32x16 (KxN), 16-bit: lane: N = lane&15, half = lane>>4;
// per-lane element j (0..15): K = 8*half + (j&7) + 16*(j>>3). Tile = 1024 B, lane-major.
// ---------------------------------------------------------------------------
__global__ void __launch_bounds__(256) k_wprep(const float* __restrict__ w1,
                                               const float* __restrict__ w2,
                                               char* __restrict__ ws) {
  const int tid  = threadIdx.x;
  const int wv   = tid >> 5, lane = tid & 31;
  const int t    = blockIdx.x * 8 + wv;          // global tile id (131072 total)
  const int half = lane >> 4, col = lane & 15;

  const float* src; int stride, row0, ncol; char* dst;
  if (t < 65536) {                               // w1: [E][nt:256][kt:32]
    int e = t >> 13, rem = t & 8191;
    int nt = rem >> 5, kt = rem & 31;
    src = w1 + (size_t)e * DMODEL * FHID;
    stride = FHID; row0 = kt * 32; ncol = nt * 16 + col;
    dst = ws + W1B_OFF + (size_t)t * 1024;
  } else {                                       // w2: [E][nt:64][kt:128]
    int u = t - 65536;
    int e = u >> 13, rem = u & 8191;
    int nt = rem >> 7, kt = rem & 127;
    src = w2 + (size_t)e * FHID * DMODEL;
    stride = DMODEL; row0 = kt * 32; ncol = nt * 16 + col;
    dst = ws + W2B_OFF + (size_t)u * 1024;
  }

  unsigned pk[8];
  #pragma unroll
  for (int jp = 0; jp < 8; ++jp) {
    int k0 = row0 + half * 8 + (jp & 3) * 2 + (jp >> 2) * 16;
    float v0 = src[(size_t)k0       * stride + ncol];
    float v1 = src[(size_t)(k0 + 1) * stride + ncol];
    pk[jp] = (unsigned)f2bf(v0) | ((unsigned)f2bf(v1) << 16);
  }
  uint4* d4 = (uint4*)(dst + lane * 32);
  d4[0] = make_uint4(pk[0], pk[1], pk[2], pk[3]);
  d4[1] = make_uint4(pk[4], pk[5], pk[6], pk[7]);
}

// ---------------------------------------------------------------------------
// Kernel 3: bin token indices per expert.
// ---------------------------------------------------------------------------
__global__ void k_bucket(char* __restrict__ ws) {
  int n = blockIdx.x * 256 + threadIdx.x;
  int e = ((const int*)(ws + EIDX_OFF))[n];
  int pos = atomicAdd(&((int*)(ws + CNT_OFF))[e], 1);
  ((int*)(ws + BKT_OFF))[e * NTOK + pos] = n;
}

// ---------------------------------------------------------------------------
// Kernel 4: fused expert FFN. Block = (expert e, group g of 32 tokens = 2
// M-tiles). Each B fragment feeds 2 WMMAs -> halved weight traffic.
// F processed in 8 chunks of 512; Y accumulated in registers.
// Phase 0 data movers: M-tile 0 via TDM gather-mode tensor_load_to_lds
// (+ in-LDS reshuffle), M-tile 1 via per-lane global_load_async_to_lds_b128.
// LDS: As (A-frag X, 2x32KB) | Hs (A-frag H chunk, 2x16KB) | tok/gate.
// ---------------------------------------------------------------------------
#define AS_OFF   0
#define HS_OFF   65536
#define TOK_SOFF (65536 + 32768)
#define GS_SOFF  (TOK_SOFF + 128)
#define SMEM_BYTES (GS_SOFF + 128)

__global__ void __launch_bounds__(256) k_moe(const char* __restrict__ ws,
                                             float* __restrict__ out) {
  extern __shared__ char smem[];
  const int e = blockIdx.x >> 6;
  const int g = blockIdx.x & 63;
  const int ce = ((const int*)(ws + CNT_OFF))[e];
  if (g * 32 >= ce) return;

  const int tid = threadIdx.x;
  const int wv  = tid >> 5, lane = tid & 31;
  const int col = lane & 15, bh = lane >> 4;

  int*   tok = (int*)(smem + TOK_SOFF);
  float* gs  = (float*)(smem + GS_SOFF);
  if (tid < 32) {
    int idx = g * 32 + tid;
    int tk = (idx < ce) ? ((const int*)(ws + BKT_OFF))[e * NTOK + idx] : -1;
    tok[tid] = tk;
    gs[tid]  = (tk >= 0) ? ((const float*)(ws + GATE_OFF))[tk] : 0.f;
  }
  __syncthreads();

  const char* xbb = ws + XB_OFF;

  // ---- phase 0a: TDM gather-mode load of M-tile 0 (16 token rows) ----
  // D# per ISA ch.8: g0 = {flags, lds_addr, global_addr, type}, g1 = dims,
  // g2/g3 = 16x 16-bit row indices. Rows land row-major [16][2048 B] at AS_OFF.
  if (wv == 0) {
    unsigned      ldsa = (unsigned)(uintptr_t)(smem + AS_OFF);
    unsigned long long ga = (unsigned long long)(uintptr_t)xbb;
    v4u g0;
    g0.x = 0x80000001u;                                 // count=1, gather, idx16
    g0.y = (unsigned)__builtin_amdgcn_readfirstlane((int)ldsa);
    g0.z = (unsigned)__builtin_amdgcn_readfirstlane((int)(unsigned)(ga & 0xffffffffull));
    g0.w = (unsigned)__builtin_amdgcn_readfirstlane(
               (int)(((unsigned)(ga >> 32) & 0x01ffffffu) | 0x80000000u)); // type=2
    v8i g1;
    g1[0] = 0x00010000;                // wg_mask=0, data_size=2B, no pad/iter
    g1[1] = (int)(1024u << 16);        // tensor_dim0 = 1024 elements
    g1[2] = (int)(2048u << 16);        // tensor_dim1 = 2048 rows
    g1[3] = (int)(1024u << 16);        // tile_dim0   = 1024 elements
    g1[4] = 16;                        // tile_dim1   = 16 gather indices
    g1[5] = 1024;                      // tensor_dim0_stride = 1024 elements
    g1[6] = 0;
    g1[7] = 0;
    int pr[8];
    #pragma unroll
    for (int q = 0; q < 8; ++q) {
      int t0 = tok[2 * q];     if (t0 < 0) t0 = 0;
      int t1 = tok[2 * q + 1]; if (t1 < 0) t1 = 0;
      pr[q] = __builtin_amdgcn_readfirstlane(t0 | (t1 << 16));
    }
    v4i g2 = {pr[0], pr[1], pr[2], pr[3]};
    v4i g3 = {pr[4], pr[5], pr[6], pr[7]};
    v8i g4 = {0, 0, 0, 0, 0, 0, 0, 0};   // extra group (clang-23 6-arg form)
    __builtin_amdgcn_tensor_load_to_lds(g0, g1, g2, g3, g4, 0);
  }

  // ---- phase 0b: M-tile 1 gathered straight into A-fragment layout ----
  #pragma unroll
  for (int i = 0; i < 8; ++i) {
    int c   = 2048 + i * 256 + tid;            // chunks of M-tile 1
    int kt  = (c >> 6) & 31;
    int ln  = (c & 63) >> 1;
    int h   = c & 1;
    int m   = 16 + (ln & 15);
    int tk  = tok[m]; if (tk < 0) tk = 0;
    int ksrc = kt * 32 + (ln >> 4) * 8 + h * 16;
    unsigned ldsa = (unsigned)(uintptr_t)(smem + AS_OFF + 32768
                                          + kt * 1024 + ln * 32 + h * 16);
    unsigned long long ga =
        (unsigned long long)(uintptr_t)(xbb + (size_t)tk * (DMODEL * 2) + ksrc * 2);
    asm volatile("global_load_async_to_lds_b128 %0, %1, off"
                 :: "v"(ldsa), "v"(ga) : "memory");
  }
  asm volatile("s_wait_asynccnt 0x0" ::: "memory");
  if (wv == 0) __builtin_amdgcn_s_wait_tensorcnt(0);
  __syncthreads();

  // ---- phase 0c: reshuffle M-tile 0 row-major -> A-fragment (in place) ----
  {
    uint4 tmp[8];
    #pragma unroll
    for (int i = 0; i < 8; ++i) {
      int c  = i * 256 + tid;
      int kt = (c >> 6) & 31, ln = (c & 63) >> 1, h = c & 1, m = ln & 15;
      tmp[i] = *(const uint4*)(smem + AS_OFF + m * 2048 + kt * 64
                               + (ln >> 4) * 16 + h * 32);
    }
    __syncthreads();
    #pragma unroll
    for (int i = 0; i < 8; ++i) {
      int c  = i * 256 + tid;
      int kt = (c >> 6) & 31, ln = (c & 63) >> 1, h = c & 1;
      *(uint4*)(smem + AS_OFF + kt * 1024 + ln * 32 + h * 16) = tmp[i];
    }
  }
  __syncthreads();

  const char* w1b = ws + W1B_OFF;
  const char* w2b = ws + W2B_OFF;

  v8f acc[2][8];
  #pragma unroll
  for (int mt = 0; mt < 2; ++mt)
    #pragma unroll
    for (int j = 0; j < 8; ++j)
      acc[mt][j] = v8f{0.f, 0.f, 0.f, 0.f, 0.f, 0.f, 0.f, 0.f};

  for (int fc = 0; fc < 8; ++fc) {
    // -- phase 1: H[32, 512] = relu(X @ W1[e][:, fc*512 : +512]) --
    #pragma unroll
    for (int ntl = 0; ntl < 4; ++ntl) {
      const int ntc = wv * 4 + ntl;            // 0..31 within chunk
      const int nt  = fc * 32 + ntc;           // 0..255 global N-tile
      v8f h0 = v8f{0.f, 0.f, 0.f, 0.f, 0.f, 0.f, 0.f, 0.f};
      v8f h1 = v8f{0.f, 0.f, 0.f, 0.f, 0.f, 0.f, 0.f, 0.f};
      const char* bt = w1b + ((size_t)(e * 256 + nt) * 32) * 1024 + lane * 32;
      const char* a0 = smem + AS_OFF + lane * 32;
      if (ntl + 1 < 4) __builtin_prefetch(bt + 32 * 1024, 0, 0);
      #pragma unroll 4
      for (int kt = 0; kt < 32; ++kt) {
        v16bf b  = *(const v16bf*)(bt + kt * 1024);
        v16bf x0 = *(const v16bf*)(a0 + kt * 1024);
        v16bf x1 = *(const v16bf*)(a0 + 32768 + kt * 1024);
        h0 = __builtin_amdgcn_wmma_f32_16x16x32_bf16(false, x0, false, b,
                                                     (short)0, h0, false, false);
        h1 = __builtin_amdgcn_wmma_f32_16x16x32_bf16(false, x1, false, b,
                                                     (short)0, h1, false, false);
      }
      // relu + scatter into Hs chunk in phase-2 A-fragment order
      const int kt2 = ntc >> 1;
      const int j   = ((ntc & 1) << 3) + (col & 7);
      char* hb = smem + HS_OFF + kt2 * 1024
               + (((col >> 3) << 4) + (bh << 3)) * 32 + j * 2;
      #pragma unroll
      for (int v = 0; v < 8; ++v) {
        *(unsigned short*)(hb + v * 32)         = f2bf(fmaxf(h0[v], 0.f));
        *(unsigned short*)(hb + 16384 + v * 32) = f2bf(fmaxf(h1[v], 0.f));
      }
    }
    __syncthreads();

    // -- phase 2: Y += H_chunk @ W2[e][fc*512 : +512, :] (persistent accs) --
    const char* at = smem + HS_OFF + lane * 32;
    const char* bbase = w2b + ((size_t)(e * 64 + wv * 8) * 128 + fc * 16) * 1024
                      + lane * 32;
    #pragma unroll 2
    for (int kt = 0; kt < 16; ++kt) {
      v16bf ha = *(const v16bf*)(at + kt * 1024);
      v16bf hc = *(const v16bf*)(at + 16384 + kt * 1024);
      #pragma unroll
      for (int ntl = 0; ntl < 8; ++ntl) {
        v16bf b = *(const v16bf*)(bbase + (size_t)ntl * 131072 + kt * 1024);
        acc[0][ntl] = __builtin_amdgcn_wmma_f32_16x16x32_bf16(
            false, ha, false, b, (short)0, acc[0][ntl], false, false);
        acc[1][ntl] = __builtin_amdgcn_wmma_f32_16x16x32_bf16(
            false, hc, false, b, (short)0, acc[1][ntl], false, false);
      }
    }
    __syncthreads();   // Hs reused next chunk
  }

  // ---- epilogue: gate-scale and scatter rows ----
  #pragma unroll
  for (int mt = 0; mt < 2; ++mt)
    #pragma unroll
    for (int ntl = 0; ntl < 8; ++ntl) {
      const int n2 = wv * 8 + ntl;
      #pragma unroll
      for (int v = 0; v < 8; ++v) {
        int m  = mt * 16 + v + (bh << 3);
        int tk = tok[m];
        if (tk >= 0)
          out[(size_t)tk * DMODEL + n2 * 16 + col] = acc[mt][ntl][v] * gs[m];
      }
    }
}

// ---------------------------------------------------------------------------
extern "C" void kernel_launch(void* const* d_in, const int* in_sizes, int n_in,
                              void* d_out, int out_size, void* d_ws, size_t ws_size,
                              hipStream_t stream) {
  const float* x    = (const float*)d_in[0];
  const int*   mask = (const int*)d_in[1];
  const float* rw   = (const float*)d_in[2];
  const float* w1   = (const float*)d_in[3];
  const float* w2   = (const float*)d_in[4];
  char*  ws  = (char*)d_ws;
  float* out = (float*)d_out;

  k_router<<<dim3(NTOK), dim3(256), 0, stream>>>(x, mask, rw, ws);
  k_wprep <<<dim3(16384), dim3(256), 0, stream>>>(w1, w2, ws);
  k_bucket<<<dim3(NTOK / 256), dim3(256), 0, stream>>>(ws);
  k_moe   <<<dim3(NEXP * 64), dim3(256), SMEM_BYTES, stream>>>(ws, out);
}
